// PTQLinear_10900626997679
// MI455X (gfx1250) — matrix operand order repaired and checked by reading, not compiled, not run on hardware
//
#include <hip/hip_runtime.h>
#include <stdint.h>

typedef __attribute__((ext_vector_type(16))) _Float16 v16h;
typedef __attribute__((ext_vector_type(8)))  _Float16 v8h;
typedef __attribute__((ext_vector_type(4)))  _Float16 v4h;
typedef __attribute__((ext_vector_type(8)))  float    v8f;

#define TOKENS 8192
#define D_IN   4096
#define D_OUT  4096

#define BM 128
#define BN 128
#define BK 64
#define LDSS 72    // padded LDS row stride in halves (64 + 8), keeps 16B alignment
#define NBUF 4     // staging buffers, prefetch distance 2

// ---------------- CDNA5 async global->LDS copy (16B per lane) ----------------
// GLOBAL_LOAD_ASYNC_TO_LDS_B128, GV mode: vdst = LDS byte address (wave-relative),
// vaddr = 64-bit global address. Tracked on ASYNCcnt, bypasses VGPRs.
__device__ __forceinline__ void async_ld16(const _Float16* g, _Float16* l) {
    uint32_t lds_addr = (uint32_t)(uintptr_t)l;
    asm volatile("global_load_async_to_lds_b128 %0, %1, off"
                 :: "v"(lds_addr), "v"(g)
                 : "memory");
}

// async loads retire in order: cnt<=N means all but the newest N have landed
#define WAIT_ASYNCCNT(n) asm volatile("s_wait_asynccnt " #n ::: "memory")

// ---------------- conversion kernels ----------------

__global__ __launch_bounds__(256) void cvt_f32_to_f16(const float4* __restrict__ src,
                                                      v4h* __restrict__ dst, int n4) {
    int stride = gridDim.x * blockDim.x;
    for (int i = blockIdx.x * blockDim.x + threadIdx.x; i < n4; i += stride) {
        float4 v = src[i];
        v4h h;
        h[0] = (_Float16)v.x; h[1] = (_Float16)v.y;
        h[2] = (_Float16)v.z; h[3] = (_Float16)v.w;
        dst[i] = h;
    }
}

__global__ __launch_bounds__(256) void cvt_i32_to_f16(const int4* __restrict__ src,
                                                      v4h* __restrict__ dst, int n4) {
    int stride = gridDim.x * blockDim.x;
    for (int i = blockIdx.x * blockDim.x + threadIdx.x; i < n4; i += stride) {
        int4 v = src[i];
        v4h h;
        h[0] = (_Float16)(float)v.x; h[1] = (_Float16)(float)v.y;
        h[2] = (_Float16)(float)v.z; h[3] = (_Float16)(float)v.w;
        dst[i] = h;
    }
}

// ---------------- WMMA GEMM, 4-deep async staging, distance-2 prefetch -------
// A: x_f16  [TOKENS][D_IN]  row-major
// B: w_f16  [D_OUT][D_IN]   row-major (= B of x @ W^T; K contiguous per N row)
// C = (A*B^T) * scales[n] + bias[n], f32

__global__ __launch_bounds__(256) void ptq_gemm_f16(const _Float16* __restrict__ A,
                                                    const _Float16* __restrict__ B,
                                                    const float* __restrict__ scales,
                                                    const float* __restrict__ bias,
                                                    float* __restrict__ C) {
    __shared__ _Float16 sA[NBUF][BM * LDSS];   // 4 x 18 KB
    __shared__ _Float16 sB[NBUF][BN * LDSS];   // 4 x 18 KB

    const int tid   = threadIdx.x;
    const int wave  = tid >> 5;
    const int lane  = tid & 31;
    const int l16   = lane & 15;
    const int lhalf = lane >> 4;      // 0 or 1
    const int waveM = wave >> 2;      // 0..1  -> 64-row slab
    const int waveN = wave & 3;       // 0..3  -> 32-col slab
    const int mBase = blockIdx.y * BM;
    const int nBase = blockIdx.x * BN;

    // staging mapping: 1024 chunks of 8 halves (16B) per tile, 4 per thread
    const int srow = tid >> 3;         // 0..31 (+32*i)
    const int scol = (tid & 7) * 8;    // half offset within the 64-wide k-chunk

    const _Float16* gA = A + (size_t)(mBase + srow) * D_IN + scol;
    const _Float16* gB = B + (size_t)(nBase + srow) * D_IN + scol;

    auto stage = [&](int buf, int kBase) {
#pragma unroll
        for (int i = 0; i < 4; ++i) {
            const int r = srow + 32 * i;
            async_ld16(gA + (size_t)(32 * i) * D_IN + kBase,
                       &sA[buf][r * LDSS + scol]);
            async_ld16(gB + (size_t)(32 * i) * D_IN + kBase,
                       &sB[buf][r * LDSS + scol]);
        }
    };

    auto load_frag = [](const _Float16* p, int hiOff) -> v16h {
        v8h lo = *(const v8h*)p;
        v8h hi = *(const v8h*)(p + hiOff);
        return __builtin_shufflevector(lo, hi, 0, 1, 2, 3, 4, 5, 6, 7,
                                       8, 9, 10, 11, 12, 13, 14, 15);
    };

    v8f acc[4][2];
#pragma unroll
    for (int tm = 0; tm < 4; ++tm)
#pragma unroll
        for (int tn = 0; tn < 2; ++tn)
            acc[tm][tn] = (v8f){0.f, 0.f, 0.f, 0.f, 0.f, 0.f, 0.f, 0.f};

    // compute one 64-wide k-chunk from one LDS buffer pair
    auto compute = [&](const _Float16* bufA, const _Float16* bufB) {
#pragma unroll
        for (int ks = 0; ks < 2; ++ks) {
            const int kk = ks * 32;
            // issue all 12 ds_load_b128 for this k-step before any WMMA
            v16h bf0 = load_frag(
                &bufB[(waveN * 32 + 0 * 16 + l16) * LDSS + kk + lhalf * 16], 8);
            v16h bf1 = load_frag(
                &bufB[(waveN * 32 + 1 * 16 + l16) * LDSS + kk + lhalf * 16], 8);
            v16h af[4];
#pragma unroll
            for (int tm = 0; tm < 4; ++tm)
                af[tm] = load_frag(
                    &bufA[(waveM * 64 + tm * 16 + l16) * LDSS + kk + lhalf * 16], 8);
#pragma unroll
            for (int tm = 0; tm < 4; ++tm) {
                acc[tm][0] = __builtin_amdgcn_wmma_f32_16x16x32_f16(
                    false, af[tm], false, bf0, (short)0, acc[tm][0], false, false);
                acc[tm][1] = __builtin_amdgcn_wmma_f32_16x16x32_f16(
                    false, af[tm], false, bf1, (short)0, acc[tm][1], false, false);
            }
        }
    };

    // prologue: fill buffers 0 and 1 (stage 1 may stay in flight)
    stage(0, 0);
    stage(1, BK);
    WAIT_ASYNCCNT(0x8);     // buffer 0 landed
    __syncthreads();

    const int NKC = D_IN / BK;   // 64

    // steady state: always stages -> branch-free body
    for (int kc = 0; kc < NKC - 2; ++kc) {
        stage((kc + 2) & (NBUF - 1), (kc + 2) * BK);
        compute(sA[kc & (NBUF - 1)], sB[kc & (NBUF - 1)]);
        // oldest stage (kc+1) must have landed; the one just issued may stay in flight
        WAIT_ASYNCCNT(0x8);
        __syncthreads();
    }

    // pipeline drain: two buffered chunks remain
    compute(sA[(NKC - 2) & (NBUF - 1)], sB[(NKC - 2) & (NBUF - 1)]);
    WAIT_ASYNCCNT(0x0);
    __syncthreads();
    compute(sA[(NKC - 1) & (NBUF - 1)], sB[(NKC - 1) & (NBUF - 1)]);

    // ---- epilogue: dequant scale + bias, f32 store ----
    // C layout: VGPR r, lanes 0-15 -> M=r, lanes 16-31 -> M=r+8; N = lane&15
#pragma unroll
    for (int tn = 0; tn < 2; ++tn) {
        const int n = nBase + waveN * 32 + tn * 16 + l16;
        const float s = scales[n];
        const float bb = bias[n];
#pragma unroll
        for (int tm = 0; tm < 4; ++tm) {
            const int m0 = mBase + waveM * 64 + tm * 16 + lhalf * 8;
#pragma unroll
            for (int r = 0; r < 8; ++r) {
                C[(size_t)(m0 + r) * D_OUT + n] = acc[tm][tn][r] * s + bb;
            }
        }
    }
}

// ---------------- host launch ----------------

extern "C" void kernel_launch(void* const* d_in, const int* in_sizes, int n_in,
                              void* d_out, int out_size, void* d_ws, size_t ws_size,
                              hipStream_t stream) {
    const float* x      = (const float*)d_in[0];
    const int*   wq     = (const int*)d_in[1];
    const float* scales = (const float*)d_in[2];
    const float* bias   = (const float*)d_in[3];
    float*       out    = (float*)d_out;

    _Float16* xh = (_Float16*)d_ws;
    _Float16* wh = (_Float16*)((char*)d_ws + (size_t)TOKENS * D_IN * sizeof(_Float16));

    cvt_f32_to_f16<<<4096, 256, 0, stream>>>((const float4*)x, (v4h*)xh,
                                             (TOKENS * D_IN) / 4);
    cvt_i32_to_f16<<<4096, 256, 0, stream>>>((const int4*)wq, (v4h*)wh,
                                             (D_OUT * D_IN) / 4);

    dim3 grid(D_OUT / BN, TOKENS / BM);
    ptq_gemm_f16<<<grid, 256, 0, stream>>>(xh, wh, scales, bias, out);
}
